// LF_Enhanced_Encoder_3753801416932
// MI455X (gfx1250) — compile-verified, matrix operand order checked by reading, not run on hardware
//
#include <hip/hip_runtime.h>
#include <hip/hip_bf16.h>

// ---------------------------------------------------------------------------
// Model dims
// ---------------------------------------------------------------------------
#define HH     512
#define NINP   300
#define KP     320     // NINP padded to multiple of 32
#define IMGD   2048
#define BB     256
#define TQn    20
#define RNDn   10
#define THISTn 40
#define Rn     36
#define G4     2048    // 4*H
#define NHIST  2560    // B*RND
#define VOCABn 11322

typedef __attribute__((ext_vector_type(16))) __bf16 v16bf;
typedef __attribute__((ext_vector_type(8)))  __bf16 v8bf;
typedef __attribute__((ext_vector_type(8)))  float  v8f;
typedef __attribute__((ext_vector_type(4)))  float  f4v;

// ---------------------------------------------------------------------------
// WMMA GEMM:  C[M,N] = X[M,K] * W[N,K]^T (+bias) (+leaky_relu)
//  - W is pre-converted bf16 (rows 16B aligned), K % 32 == 0 always.
//  - A is f32 (a_bf16=0) or bf16 (a_bf16=1); optionally row-gathered via gidx.
//  - A 64x32 slab staged through LDS (bf16) once per block per k-step,
//    shared by all 8 waves; each wave computes a 64x32 C tile (8 WMMAs/step).
// Fragment layout per CDNA5 ISA 7.12.2 (16-bit A 16x32; B symmetric; C 8xf32)
// ---------------------------------------------------------------------------
__device__ __forceinline__ v16bf bfrag(const __bf16* __restrict__ base, int hi) {
  v8bf c1 = *(const v8bf*)(base + hi * 8);
  v8bf c2 = *(const v8bf*)(base + 16 + hi * 8);
  return __builtin_shufflevector(c1, c2, 0, 1, 2, 3, 4, 5, 6, 7,
                                 8, 9, 10, 11, 12, 13, 14, 15);
}

__global__ __launch_bounds__(256) void gemm_wmma(
    const void* __restrict__ A, long lda, int a_bf16,
    const int* __restrict__ gidx, int gstride, int goff,
    const __bf16* __restrict__ W, long ldw,
    const float* __restrict__ bias,
    float* __restrict__ C, long ldc,
    int M, int N, int K, int act) {
  __shared__ __align__(16) __bf16 As[64 * 32];   // A slab, bf16, row-major

  const int tid   = threadIdx.x;
  const int wave  = tid >> 5;
  const int lane  = tid & 31;
  const int r     = lane & 15;
  const int hi    = lane >> 4;
  const int tileN  = blockIdx.x * 256 + wave * 32;
  const int tileM0 = blockIdx.y * 64;

  // two B rows per lane (clamped; stores are guarded)
  const int coln0 = tileN + r;
  const int coln1 = tileN + 16 + r;
  const __bf16* wrow0 = W + (size_t)(coln0 < N ? coln0 : N - 1) * ldw;
  const __bf16* wrow1 = W + (size_t)(coln1 < N ? coln1 : N - 1) * ldw;

  // staging: each thread owns 8 contiguous cols of one slab row
  const int srow = tid >> 2;          // 0..63
  const int scol = (tid & 3) * 8;     // 0,8,16,24
  const float*  rpf;
  const __bf16* rpb;
  {
    int grow = tileM0 + srow;
    int rr   = grow < M ? grow : 0;
    long ridx = gidx ? (long)gidx[(size_t)rr * gstride + goff] : (long)rr;
    rpf = (const float*)A  + ridx * lda;
    rpb = (const __bf16*)A + ridx * lda;
  }

  v8f acc[8] = {};
  for (int k0 = 0; k0 < K; k0 += 32) {
    // ---- stage A slab into LDS ----
    v8bf sb;
    if (a_bf16) {
      sb = *(const v8bf*)(rpb + k0 + scol);
    } else {
      const f4v u = *(const f4v*)(rpf + k0 + scol);
      const f4v v = *(const f4v*)(rpf + k0 + scol + 4);
#pragma unroll
      for (int j = 0; j < 4; ++j) { sb[j] = (__bf16)u[j]; sb[j + 4] = (__bf16)v[j]; }
    }
    *(v8bf*)(As + srow * 32 + scol) = sb;
    __syncthreads();

    // ---- B fragments from global (bf16) ----
    v16bf b0 = bfrag(wrow0 + k0, hi);
    v16bf b1 = bfrag(wrow1 + k0, hi);

    // ---- A fragments from LDS ----
    v16bf a[4];
#pragma unroll
    for (int mi = 0; mi < 4; ++mi)
      a[mi] = bfrag(As + (mi * 16 + r) * 32, hi);
    __syncthreads();

#pragma unroll
    for (int mi = 0; mi < 4; ++mi) {
      acc[mi * 2 + 0] = __builtin_amdgcn_wmma_f32_16x16x32_bf16(
          false, a[mi], false, b0, (short)0, acc[mi * 2 + 0], false, false);
      acc[mi * 2 + 1] = __builtin_amdgcn_wmma_f32_16x16x32_bf16(
          false, a[mi], false, b1, (short)0, acc[mi * 2 + 1], false, false);
    }
  }

#pragma unroll
  for (int ni = 0; ni < 2; ++ni) {
    const int coln = tileN + ni * 16 + r;
    if (coln < N) {
      const float bval = bias ? bias[coln] : 0.0f;
#pragma unroll
      for (int mi = 0; mi < 4; ++mi) {
#pragma unroll
        for (int v = 0; v < 8; ++v) {
          int row = tileM0 + mi * 16 + hi * 8 + v;
          if (row < M) {
            float x = acc[mi * 2 + ni][v] + bval;
            if (act) x = x > 0.0f ? x : 0.01f * x;
            C[(size_t)row * ldc + coln] = x;
          }
        }
      }
    }
  }
}

// ---------------------------------------------------------------------------
// One-time f32 -> bf16 weight conversion (with K padding)
// ---------------------------------------------------------------------------
__global__ void cvt_w(__bf16* __restrict__ dst, const float* __restrict__ src,
                      int rows, int Ksrc, int Kdst) {
  int i = blockIdx.x * blockDim.x + threadIdx.x;
  if (i >= rows * Kdst) return;
  int rI = i / Kdst, k = i - rI * Kdst;
  dst[i] = (__bf16)(k < Ksrc ? src[(size_t)rI * Ksrc + k] : 0.0f);
}

// ---------------------------------------------------------------------------
// Elementwise / wave helper kernels
// ---------------------------------------------------------------------------
__global__ void zero_k(float* __restrict__ p, int n) {
  int i = blockIdx.x * blockDim.x + threadIdx.x;
  if (i < n) p[i] = 0.0f;
}

// gather embedding rows, zero-padding cols NINP..KP
__global__ void gather_emb_pad(float* __restrict__ out, const float* __restrict__ emb,
                               const int* __restrict__ tok, int T, int tokstride,
                               int nrows) {
  int i = blockIdx.x * blockDim.x + threadIdx.x;
  if (i >= nrows * KP) return;
  int row = i / KP, k = i - row * KP;
  int b = row / T, t = row - b * T;
  int id = tok[(size_t)b * tokstride + t];
  out[i] = (k < NINP) ? emb[(size_t)id * NINP + k] : 0.0f;
}

__global__ void copy2d(float* __restrict__ dst, long dld,
                       const float* __restrict__ src, long sld, int B, int n) {
  int i = blockIdx.x * blockDim.x + threadIdx.x;
  if (i >= B * n) return;
  int b = i / n, j = i - b * n;
  dst[(size_t)b * dld + j] = src[(size_t)b * sld + j];
}

// LSTM gate pointwise: g = g1 + g2 + bih + bhh ; split i,f,g,o
__global__ void lstm_gate(const float* __restrict__ g1, const float* __restrict__ g2,
                          const float* __restrict__ bih, const float* __restrict__ bhh,
                          float* __restrict__ h, float* __restrict__ c,
                          float* __restrict__ seq, int T, int t,
                          float* __restrict__ enc, const int* __restrict__ lens,
                          int lstride, int NB) {
  int i = blockIdx.x * blockDim.x + threadIdx.x;
  if (i >= NB * HH) return;
  int m = i >> 9, j = i & 511;
  size_t base = (size_t)m * G4 + j;
  float gi = g1[base]        + g2[base]        + bih[j]        + bhh[j];
  float gf = g1[base + 512]  + g2[base + 512]  + bih[512 + j]  + bhh[512 + j];
  float gg = g1[base + 1024] + g2[base + 1024] + bih[1024 + j] + bhh[1024 + j];
  float go = g1[base + 1536] + g2[base + 1536] + bih[1536 + j] + bhh[1536 + j];
  float ig = 1.0f / (1.0f + expf(-gi));
  float fg = 1.0f / (1.0f + expf(-gf));
  float og = 1.0f / (1.0f + expf(-go));
  float cn = fg * c[i] + ig * tanhf(gg);
  float hn = og * tanhf(cn);
  c[i] = cn; h[i] = hn;
  if (seq) seq[((size_t)m * T + t) * HH + j] = hn;
  if (enc && t == lens[(size_t)m * lstride] - 1) enc[(size_t)m * HH + j] = hn;
}

// s[m] = wb + sum_j wall[j]*tanh( mode? fv*qv : fv+qv(+cv) ), b = m/T. One wave/row.
__global__ void score_rows(float* __restrict__ s, const float* __restrict__ fv,
                           const float* __restrict__ qv, const float* __restrict__ cv,
                           const float* __restrict__ wall, const float* __restrict__ wb,
                           int M, int T, int mode) {
  int wid = (blockIdx.x * blockDim.x + threadIdx.x) >> 5;
  int lane = threadIdx.x & 31;
  if (wid >= M) return;
  int b = wid / T;
  float sum = 0.0f;
  for (int j = lane; j < HH; j += 32) {
    float fvv = fv[(size_t)wid * HH + j];
    float qvv = qv[(size_t)b * HH + j];
    float v;
    if (mode) v = fvv * qvv;
    else { v = fvv + qvv; if (cv) v += cv[(size_t)b * HH + j]; }
    sum += wall[j] * tanhf(v);
  }
  for (int o = 16; o; o >>= 1) sum += __shfl_xor(sum, o, 32);
  if (lane == 0) s[wid] = sum + wb[0];
}

// softmax over rows of length T; if lens != null apply reference mask
// mask = (x==0) | (t >= len) -> -100000 before softmax. One wave/row.
__global__ void softmax_rows(float* __restrict__ w, const float* __restrict__ s,
                             const int* __restrict__ lens, int lstride, int M, int T) {
  int wid = (blockIdx.x * blockDim.x + threadIdx.x) >> 5;
  int lane = threadIdx.x & 31;
  if (wid >= M) return;
  int len = lens ? lens[(size_t)wid * lstride] : T;
  float mx = -3.0e38f;
  for (int t = lane; t < T; t += 32) {
    float x = s[(size_t)wid * T + t];
    if (lens && (x == 0.0f || t >= len)) x = -100000.0f;
    mx = fmaxf(mx, x);
  }
  for (int o = 16; o; o >>= 1) mx = fmaxf(mx, __shfl_xor(mx, o, 32));
  float sm = 0.0f;
  for (int t = lane; t < T; t += 32) {
    float x = s[(size_t)wid * T + t];
    if (lens && (x == 0.0f || t >= len)) x = -100000.0f;
    sm += expf(x - mx);
  }
  for (int o = 16; o; o >>= 1) sm += __shfl_xor(sm, o, 32);
  float inv = 1.0f / sm;
  for (int t = lane; t < T; t += 32) {
    float x = s[(size_t)wid * T + t];
    if (lens && (x == 0.0f || t >= len)) x = -100000.0f;
    w[(size_t)wid * T + t] = expf(x - mx) * inv;
  }
}

// out[b, 0..D) (row stride ostride) = sum_t w[b,t] * E[b,t,d]
__global__ void wsum_k(float* __restrict__ out, long ostride,
                       const float* __restrict__ w, const float* __restrict__ E,
                       int B, int T, int D) {
  int i = blockIdx.x * blockDim.x + threadIdx.x;
  if (i >= B * D) return;
  int b = i / D, d = i - b * D;
  float a = 0.0f;
  for (int t = 0; t < T; ++t)
    a += w[(size_t)b * T + t] * E[((size_t)b * T + t) * D + d];
  out[(size_t)b * ostride + d] = a;
}

// tiny dense layers (Nout in {1,2,3}): one wave per (b,n) dot of length K
__global__ void small_lin(float* __restrict__ out, const float* __restrict__ X,
                          const float* __restrict__ W, const float* __restrict__ bias,
                          int Brows, int K, int Nout, int act) {
  int wid = (blockIdx.x * blockDim.x + threadIdx.x) >> 5;
  int lane = threadIdx.x & 31;
  if (wid >= Brows * Nout) return;
  int b = wid / Nout, n = wid - b * Nout;
  float sum = 0.0f;
  for (int k = lane; k < K; k += 32) sum += X[(size_t)b * K + k] * W[(size_t)n * K + k];
  for (int o = 16; o; o >>= 1) sum += __shfl_xor(sum, o, 32);
  if (lane == 0) {
    float x = sum + bias[n];
    if (act) x = x > 0.0f ? x : 0.01f * x;
    out[(size_t)b * Nout + n] = x;
  }
}

// ---------------------------------------------------------------------------
// Host side
// ---------------------------------------------------------------------------
static inline int cdiv(long a, long b) { return (int)((a + b - 1) / b); }

extern "C" void kernel_launch(void* const* d_in, const int* in_sizes, int n_in,
                              void* d_out, int out_size, void* d_ws, size_t ws_size,
                              hipStream_t stream) {
  (void)in_sizes; (void)n_in; (void)out_size; (void)ws_size;
  auto f32 = [&](int i) { return (const float*)d_in[i]; };
  auto i32 = [&](int i) { return (const int*)d_in[i]; };

  // params in setup_inputs() insertion order
  const float* emb   = f32(0);
  const float* qWih = f32(1),  *qWhh = f32(2),  *qbih = f32(3),  *qbhh = f32(4);
  const float* cWih = f32(5),  *cWhh = f32(6),  *cbih = f32(7),  *cbhh = f32(8);
  const float* hWih = f32(9),  *hWhh = f32(10), *hbih = f32(11), *hbhh = f32(12);
  const float *Wq2W=f32(13),*Wq2b=f32(14), *Wi2W=f32(15),*Wi2b=f32(16);
  const float *Wall2W=f32(17),*Wall2b=f32(18);
  const float *Wqs3W=f32(19),*Wqs3b=f32(20), *Wcs3W=f32(21),*Wcs3b=f32(22);
  const float *Wc3W=f32(23),*Wc3b=f32(24), *Wall3W=f32(25),*Wall3b=f32(26);
  const float *c2cW=f32(27),*c2cb=f32(28);
  const float *Wqs5W=f32(29),*Wqs5b=f32(30), *Wcs5W=f32(31),*Wcs5b=f32(32);
  const float *Wq5W=f32(33),*Wq5b=f32(34), *Wall5W=f32(35),*Wall5b=f32(36);
  const float *q2qW=f32(37),*q2qb=f32(38);
  const float *Wq1W=f32(39),*Wq1b=f32(40), *Wh1W=f32(41),*Wh1b=f32(42);
  const float *Wqh1W=f32(43),*Wqh1b=f32(44);
  const float *Wc4W=f32(45),*Wc4b=f32(46), *Wi4W=f32(47),*Wi4b=f32(48);
  const float *Wall4W=f32(49),*Wall4b=f32(50);
  const float *fu1W=f32(51),*fu1b=f32(52), *fu2W=f32(53),*fu2b=f32(54);
  const float *fu3W=f32(55),*fu3b=f32(56);
  const float *qr1W=f32(57),*qr1b=f32(58), *qr2W=f32(59),*qr2b=f32(60);
  const float *qm1W=f32(61),*qm1b=f32(62), *qm2W=f32(63),*qm2b=f32(64);
  const float* img  = f32(65);
  const int*   ques = i32(66);
  const int*   hist = i32(67);
  const int*   qlen = i32(68);
  const int*   hlen = i32(69);

  // workspace bump allocator (16B aligned slots)
  char* wp = (char*)d_ws;
  auto allocf = [&](size_t n) { float* p = (float*)wp; wp += ((n * 4 + 15) & ~(size_t)15); return p; };
  auto allocb = [&](size_t n) { __bf16* p = (__bf16*)wp; wp += ((n * 2 + 15) & ~(size_t)15); return p; };

  float* ques_embed = allocf((size_t)BB * TQn * KP);
  float* cap_emb    = allocf((size_t)BB * THISTn * KP);
  float* q_output   = allocf((size_t)BB * TQn * HH);
  float* c_output   = allocf((size_t)BB * THISTn * HH);
  float* his_feat   = allocf((size_t)NHIST * HH);
  float* ques_enc   = allocf((size_t)BB * HH);
  float* cap_enc    = allocf((size_t)BB * HH);
  float* hstate     = allocf((size_t)NHIST * HH);
  float* cstate     = allocf((size_t)NHIST * HH);
  float* g1         = allocf((size_t)NHIST * G4);
  float* g2         = allocf((size_t)NHIST * G4);
  float* qh         = allocf((size_t)BB * HH);
  float* hhb        = allocf((size_t)NHIST * HH);
  float* sbuf       = allocf((size_t)BB * THISTn);
  float* wbuf       = allocf((size_t)BB * THISTn);
  float* qatt       = allocf((size_t)BB * HH);
  float* fv         = allocf((size_t)BB * THISTn * HH);
  float* qv         = allocf((size_t)BB * HH);
  float* cv         = allocf((size_t)BB * HH);
  float* ctx        = allocf((size_t)BB * KP);
  float* acf0       = allocf((size_t)BB * HH);
  float* aqf0       = allocf((size_t)BB * HH);
  float* hac        = allocf((size_t)BB * HH);
  float* aqf        = allocf((size_t)BB * 2 * HH);
  float* acf        = allocf((size_t)BB * 2 * HH);
  float* iv         = allocf((size_t)BB * Rn * HH);
  float* cated      = allocf((size_t)BB * 2 * IMGD);
  float* qref_in    = allocf((size_t)BB * 2 * HH);
  float* qrefh      = allocf((size_t)BB * HH);
  float* fimg       = allocf((size_t)BB * IMGD);
  float* fusedvec   = allocf((size_t)BB * (3 * HH + IMGD));
  float* fused      = allocf((size_t)BB * 3 * HH);
  float* qmh        = allocf((size_t)BB * HH);

  auto g1d = [&](long n) { return dim3((unsigned)cdiv(n, 256)); };
  auto gwv = [&](long waves) { return dim3((unsigned)cdiv(waves, 8)); };

  // one-time bf16 weight conversion (K=300 padded to 320)
  auto cvtw = [&](const float* src, int rows, int Ksrc, int Kdst) {
    __bf16* d = allocb((size_t)rows * Kdst);
    cvt_w<<<g1d((long)rows * Kdst), 256, 0, stream>>>(d, src, rows, Ksrc, Kdst);
    return d;
  };
  __bf16* embH  = cvtw(emb,  VOCABn, NINP, KP);
  __bf16* qWihH = cvtw(qWih, G4, NINP, KP);
  __bf16* cWihH = cvtw(cWih, G4, NINP, KP);
  __bf16* hWihH = cvtw(hWih, G4, NINP, KP);
  __bf16* qWhhH = cvtw(qWhh, G4, HH, HH);
  __bf16* cWhhH = cvtw(cWhh, G4, HH, HH);
  __bf16* hWhhH = cvtw(hWhh, G4, HH, HH);
  __bf16* Wq1H  = cvtw(Wq1W, HH, HH, HH);
  __bf16* Wh1H  = cvtw(Wh1W, HH, HH, HH);
  __bf16* Wc3H  = cvtw(Wc3W, HH, HH, HH);
  __bf16* Wqs3H = cvtw(Wqs3W, HH, HH, HH);
  __bf16* Wcs3H = cvtw(Wcs3W, HH, HH, HH);
  __bf16* Wq5H  = cvtw(Wq5W, HH, HH, HH);
  __bf16* Wqs5H = cvtw(Wqs5W, HH, HH, HH);
  __bf16* Wcs5H = cvtw(Wcs5W, HH, HH, HH);
  __bf16* c2cH  = cvtw(c2cW, HH, NINP, KP);
  __bf16* q2qH  = cvtw(q2qW, HH, NINP, KP);
  __bf16* Wq2H  = cvtw(Wq2W, HH, 2 * HH, 2 * HH);
  __bf16* Wc4H  = cvtw(Wc4W, HH, 2 * HH, 2 * HH);
  __bf16* Wi2H  = cvtw(Wi2W, HH, IMGD, IMGD);
  __bf16* Wi4H  = cvtw(Wi4W, HH, IMGD, IMGD);
  __bf16* qr1H  = cvtw(qr1W, HH, 2 * HH, 2 * HH);
  __bf16* qm1H  = cvtw(qm1W, HH, 2 * HH, 2 * HH);
  const int FD = 3 * HH + IMGD; // 3584
  __bf16* fu1H  = cvtw(fu1W, HH, FD, FD);
  __bf16* fu2H  = cvtw(fu2W, HH, FD, FD);
  __bf16* fu3H  = cvtw(fu3W, HH, FD, FD);

  auto gemm = [&](const void* A, long lda, int abf, const int* gidx, int gs, int go,
                  const __bf16* W, long ldw, const float* bias,
                  float* C, long ldc, int M, int N, int K, int act) {
    dim3 grd(cdiv(N, 256), cdiv(M, 64));
    gemm_wmma<<<grd, dim3(256), 0, stream>>>(A, lda, abf, gidx, gs, go, W, ldw, bias,
                                             C, ldc, M, N, K, act);
  };

  // ---- embeddings (hist embeddings are gather-fused into the GEMM) ----
  gather_emb_pad<<<g1d((long)BB * TQn * KP), 256, 0, stream>>>(ques_embed, emb, ques, TQn, TQn, BB * TQn);
  gather_emb_pad<<<g1d((long)BB * THISTn * KP), 256, 0, stream>>>(cap_emb, emb, hist, THISTn, RNDn * THISTn, BB * THISTn);

  // ---- ques LSTM ----
  zero_k<<<g1d(BB * HH), 256, 0, stream>>>(hstate, BB * HH);
  zero_k<<<g1d(BB * HH), 256, 0, stream>>>(cstate, BB * HH);
  for (int t = 0; t < TQn; ++t) {
    gemm(ques_embed + (size_t)t * KP, (long)TQn * KP, 0, nullptr, 0, 0,
         qWihH, KP, nullptr, g1, G4, BB, G4, KP, 0);
    gemm(hstate, HH, 0, nullptr, 0, 0, qWhhH, HH, nullptr, g2, G4, BB, G4, HH, 0);
    lstm_gate<<<g1d(BB * HH), 256, 0, stream>>>(g1, g2, qbih, qbhh, hstate, cstate,
                                                q_output, TQn, t, ques_enc, qlen, 1, BB);
  }
  // ---- cap LSTM ----
  zero_k<<<g1d(BB * HH), 256, 0, stream>>>(hstate, BB * HH);
  zero_k<<<g1d(BB * HH), 256, 0, stream>>>(cstate, BB * HH);
  for (int t = 0; t < THISTn; ++t) {
    gemm(cap_emb + (size_t)t * KP, (long)THISTn * KP, 0, nullptr, 0, 0,
         cWihH, KP, nullptr, g1, G4, BB, G4, KP, 0);
    gemm(hstate, HH, 0, nullptr, 0, 0, cWhhH, HH, nullptr, g2, G4, BB, G4, HH, 0);
    lstm_gate<<<g1d(BB * HH), 256, 0, stream>>>(g1, g2, cbih, cbhh, hstate, cstate,
                                                c_output, THISTn, t, cap_enc, hlen, RNDn, BB);
  }
  // ---- hist LSTM (A gathered straight from the bf16 embedding table) ----
  zero_k<<<g1d(NHIST * HH), 256, 0, stream>>>(hstate, NHIST * HH);
  zero_k<<<g1d(NHIST * HH), 256, 0, stream>>>(cstate, NHIST * HH);
  for (int t = 0; t < THISTn; ++t) {
    gemm(embH, KP, 1, hist, THISTn, t, hWihH, KP, nullptr, g1, G4, NHIST, G4, KP, 0);
    gemm(hstate, HH, 0, nullptr, 0, 0, hWhhH, HH, nullptr, g2, G4, NHIST, G4, HH, 0);
    lstm_gate<<<g1d(NHIST * HH), 256, 0, stream>>>(g1, g2, hbih, hbhh, hstate, cstate,
                                                   nullptr, 0, t, his_feat, hlen, 1, NHIST);
  }

  // ---- q attends over history rounds ----
  gemm(ques_enc, HH, 0, nullptr, 0, 0, Wq1H, HH, Wq1b, qh, HH, BB, HH, HH, 0);
  gemm(his_feat, HH, 0, nullptr, 0, 0, Wh1H, HH, Wh1b, hhb, HH, NHIST, HH, HH, 0);
  score_rows<<<gwv(NHIST), 256, 0, stream>>>(sbuf, hhb, qh, nullptr, Wqh1W, Wqh1b, NHIST, RNDn, 0);
  softmax_rows<<<gwv(BB), 256, 0, stream>>>(wbuf, sbuf, nullptr, 0, BB, RNDn);
  wsum_k<<<g1d(BB * HH), 256, 0, stream>>>(qatt, HH, wbuf, his_feat, BB, RNDn, HH);

  // ---- attention over caption sequence (shared fv = Wc3 * c_output) ----
  gemm(c_output, HH, 0, nullptr, 0, 0, Wc3H, HH, Wc3b, fv, HH, BB * THISTn, HH, HH, 0);
  gemm(ques_enc, HH, 0, nullptr, 0, 0, Wqs3H, HH, Wqs3b, qv, HH, BB, HH, HH, 0);
  gemm(cap_enc,  HH, 0, nullptr, 0, 0, Wcs3H, HH, Wcs3b, cv, HH, BB, HH, HH, 0);
  score_rows<<<gwv(BB * THISTn), 256, 0, stream>>>(sbuf, fv, qv, cv, Wall3W, Wall3b, BB * THISTn, THISTn, 0);
  softmax_rows<<<gwv(BB), 256, 0, stream>>>(wbuf, sbuf, hlen, RNDn, BB, THISTn);
  wsum_k<<<g1d(BB * KP), 256, 0, stream>>>(ctx, KP, wbuf, cap_emb, BB, THISTn, KP);
  gemm(ctx, KP, 0, nullptr, 0, 0, c2cH, KP, c2cb, acf0, HH, BB, HH, KP, 0);
  gemm(qatt, HH, 0, nullptr, 0, 0, Wqs3H, HH, Wqs3b, qv, HH, BB, HH, HH, 0);
  score_rows<<<gwv(BB * THISTn), 256, 0, stream>>>(sbuf, fv, qv, cv, Wall3W, Wall3b, BB * THISTn, THISTn, 0);
  softmax_rows<<<gwv(BB), 256, 0, stream>>>(wbuf, sbuf, hlen, RNDn, BB, THISTn);
  wsum_k<<<g1d(BB * KP), 256, 0, stream>>>(ctx, KP, wbuf, cap_emb, BB, THISTn, KP);
  gemm(ctx, KP, 0, nullptr, 0, 0, c2cH, KP, c2cb, hac, HH, BB, HH, KP, 0);

  // ---- attention over question sequence ----
  gemm(q_output, HH, 0, nullptr, 0, 0, Wq5H, HH, Wq5b, fv, HH, BB * TQn, HH, HH, 0);
  gemm(cap_enc,  HH, 0, nullptr, 0, 0, Wqs5H, HH, Wqs5b, qv, HH, BB, HH, HH, 0);
  gemm(ques_enc, HH, 0, nullptr, 0, 0, Wcs5H, HH, Wcs5b, cv, HH, BB, HH, HH, 0);
  score_rows<<<gwv(BB * TQn), 256, 0, stream>>>(sbuf, fv, qv, cv, Wall5W, Wall5b, BB * TQn, TQn, 0);
  softmax_rows<<<gwv(BB), 256, 0, stream>>>(wbuf, sbuf, qlen, 1, BB, TQn);
  wsum_k<<<g1d(BB * KP), 256, 0, stream>>>(ctx, KP, wbuf, ques_embed, BB, TQn, KP);
  gemm(ctx, KP, 0, nullptr, 0, 0, q2qH, KP, q2qb, aqf0, HH, BB, HH, KP, 0);

  // ---- concats ----
  copy2d<<<g1d(BB * HH), 256, 0, stream>>>(aqf,       2 * HH, ques_enc, HH, BB, HH);
  copy2d<<<g1d(BB * HH), 256, 0, stream>>>(aqf + HH,  2 * HH, aqf0,     HH, BB, HH);
  copy2d<<<g1d(BB * HH), 256, 0, stream>>>(acf,       2 * HH, cap_enc,  HH, BB, HH);
  copy2d<<<g1d(BB * HH), 256, 0, stream>>>(acf + HH,  2 * HH, acf0,     HH, BB, HH);

  // ---- image attention (q path then c path) into cated[b, 2, IMGD] ----
  gemm(aqf, 2 * HH, 0, nullptr, 0, 0, Wq2H, 2 * HH, Wq2b, qv, HH, BB, HH, 2 * HH, 0);
  gemm(img, IMGD, 0, nullptr, 0, 0, Wi2H, IMGD, Wi2b, iv, HH, BB * Rn, HH, IMGD, 0);
  score_rows<<<gwv(BB * Rn), 256, 0, stream>>>(sbuf, iv, qv, nullptr, Wall2W, Wall2b, BB * Rn, Rn, 1);
  softmax_rows<<<gwv(BB), 256, 0, stream>>>(wbuf, sbuf, nullptr, 0, BB, Rn);
  wsum_k<<<g1d(BB * IMGD), 256, 0, stream>>>(cated, 2 * IMGD, wbuf, img, BB, Rn, IMGD);
  gemm(acf, 2 * HH, 0, nullptr, 0, 0, Wc4H, 2 * HH, Wc4b, qv, HH, BB, HH, 2 * HH, 0);
  gemm(img, IMGD, 0, nullptr, 0, 0, Wi4H, IMGD, Wi4b, iv, HH, BB * Rn, HH, IMGD, 0);
  score_rows<<<gwv(BB * Rn), 256, 0, stream>>>(sbuf, iv, qv, nullptr, Wall4W, Wall4b, BB * Rn, Rn, 1);
  softmax_rows<<<gwv(BB), 256, 0, stream>>>(wbuf, sbuf, nullptr, 0, BB, Rn);
  wsum_k<<<g1d(BB * IMGD), 256, 0, stream>>>(cated + IMGD, 2 * IMGD, wbuf, img, BB, Rn, IMGD);

  // ---- q_gs gate over {q_img, c_img} ----
  copy2d<<<g1d(BB * HH), 256, 0, stream>>>(qref_in,      2 * HH, aqf0, HH, BB, HH);
  copy2d<<<g1d(BB * HH), 256, 0, stream>>>(qref_in + HH, 2 * HH, hac,  HH, BB, HH);
  gemm(qref_in, 2 * HH, 0, nullptr, 0, 0, qr1H, 2 * HH, qr1b, qrefh, HH, BB, HH, 2 * HH, 1);
  small_lin<<<gwv(BB * 2), 256, 0, stream>>>(sbuf, qrefh, qr2W, qr2b, BB, HH, 2, 1);
  softmax_rows<<<gwv(BB), 256, 0, stream>>>(wbuf, sbuf, nullptr, 0, BB, 2);
  wsum_k<<<g1d(BB * IMGD), 256, 0, stream>>>(fimg, IMGD, wbuf, cated, BB, 2, IMGD);

  // ---- fusion ----
  copy2d<<<g1d(BB * 2 * HH), 256, 0, stream>>>(fusedvec,            FD, aqf,  2 * HH, BB, 2 * HH);
  copy2d<<<g1d(BB * IMGD),   256, 0, stream>>>(fusedvec + 2 * HH,   FD, fimg, IMGD,   BB, IMGD);
  copy2d<<<g1d(BB * HH),     256, 0, stream>>>(fusedvec + 2 * HH + IMGD, FD, qatt, HH, BB, HH);
  gemm(fusedvec, FD, 0, nullptr, 0, 0, fu1H, FD, fu1b, fused,          3 * HH, BB, HH, FD, 1);
  gemm(fusedvec, FD, 0, nullptr, 0, 0, fu2H, FD, fu2b, fused + HH,     3 * HH, BB, HH, FD, 1);
  gemm(fusedvec, FD, 0, nullptr, 0, 0, fu3H, FD, fu3b, fused + 2 * HH, 3 * HH, BB, HH, FD, 1);

  // ---- q_multi gate and final blend ----
  gemm(aqf, 2 * HH, 0, nullptr, 0, 0, qm1H, 2 * HH, qm1b, qmh, HH, BB, HH, 2 * HH, 1);
  small_lin<<<gwv(BB * 3), 256, 0, stream>>>(sbuf, qmh, qm2W, qm2b, BB, HH, 3, 1);
  softmax_rows<<<gwv(BB), 256, 0, stream>>>(wbuf, sbuf, nullptr, 0, BB, 3);
  wsum_k<<<g1d(BB * HH), 256, 0, stream>>>((float*)d_out, HH, wbuf, fused, BB, 3, HH);
}
